// Attention_11252814316295
// MI455X (gfx1250) — matrix operand order repaired and checked by reading, not compiled, not run on hardware
//
#include <hip/hip_runtime.h>
#include <hip/hip_bf16.h>
#include <math.h>

// ---------------------------------------------------------------------------
// Problem constants (from reference): B=4, T=1024, C=768, H=12, D=64, K=10
// ---------------------------------------------------------------------------
#define BB 4
#define TT 1024
#define CC 768
#define HH 12
#define DD 64
#define KTOP 10

typedef __attribute__((ext_vector_type(16))) _Float16 v16h;
typedef __attribute__((ext_vector_type(8)))  _Float16 v8h;
typedef __attribute__((ext_vector_type(8)))  float    v8f;

// D = A(16x32 f16) * B(32x16 f16) + C(16x16 f32)   -- codegen-confirmed builtin
__device__ __forceinline__ v8f wmma16x16x32(v16h a, v16h b, v8f c) {
    return __builtin_amdgcn_wmma_f32_16x16x32_f16(
        /*neg_a=*/false, a, /*neg_b=*/false, b,
        /*c_mod=*/(short)0, c, /*reuse_a=*/false, /*reuse_b=*/false);
}

// A-operand (16x32 f16): lane%16 = M row; lane/16 picks K-halves.
// elements 0..15 -> K = half*16 + e (contiguous, matches B)
__device__ __forceinline__ v16h load_a(const _Float16* lo, const _Float16* hi) {
    v8h l = *(const v8h*)lo;
    v8h h = *(const v8h*)hi;
    v16h a;
#pragma unroll
    for (int e = 0; e < 8; ++e) { a[e] = l[e]; a[e + 8] = h[e]; }
    return a;
}

// Async global->LDS copy of 16 bytes per lane (GVS form: SGPR base + VGPR off).
// Tracked by ASYNCcnt; drain with s_wait_asynccnt 0 before the LDS barrier.
__device__ __forceinline__ void async_copy_b128(unsigned lds_addr, unsigned goff,
                                                unsigned long long gbase) {
    asm volatile("global_load_async_to_lds_b128 %0, %1, %2"
                 :: "v"(lds_addr), "v"(goff), "s"(gbase) : "memory");
}
__device__ __forceinline__ void async_drain() {
    asm volatile("s_wait_asynccnt 0" ::: "memory");
}

__device__ __forceinline__ float wred_max(float v) {
#pragma unroll
    for (int o = 1; o < 32; o <<= 1) v = fmaxf(v, __shfl_xor(v, o, 32));
    return v;
}
__device__ __forceinline__ float wred_sum(float v) {
#pragma unroll
    for (int o = 1; o < 32; o <<= 1) v += __shfl_xor(v, o, 32);
    return v;
}

// ---------------------------------------------------------------------------
// Kernel: f32 -> f16 convert (plain)
// ---------------------------------------------------------------------------
__global__ void cvt_f16(const float* __restrict__ src, _Float16* __restrict__ dst, int n) {
    int i = blockIdx.x * blockDim.x + threadIdx.x;
    if (i < n) dst[i] = (_Float16)src[i];
}

// Kernel: f32 [r,c] -> f16 transposed [c,r]  (so WMMA B loads are contiguous)
__global__ void cvt_f16_T(const float* __restrict__ src, _Float16* __restrict__ dst,
                          int rows, int cols) {
    int i = blockIdx.x * blockDim.x + threadIdx.x;
    if (i < rows * cols) {
        int r = i / cols, c = i - r * cols;
        dst[c * rows + r] = (_Float16)src[i];
    }
}

// ---------------------------------------------------------------------------
// Fused GEMM:  Y[4096,768] = X[4096,768] * W[768,768] + bias
//   X  : f16 row-major;  WT : f16 transposed [n][k]
//   mode 0: write f16 [B,H,T,D] (q,k) | mode 2: f16 [B,H,D,T] (V^T)
//   mode 3: write f32 row-major [4096,768] (final projection into d_out)
// Block = 128 (4 waves). A-strip (16x768, 24 KB, contiguous) is staged into
// LDS with async DMA and shared by all 4 waves; each wave owns a 16x64 strip
// of output (4 accumulators -> 4 WMMAs per A fetch).
// grid = (4096/16, 768/256) = (256, 3)
// ---------------------------------------------------------------------------
__global__ __launch_bounds__(128) void gemm_f16(
    const _Float16* __restrict__ X, const _Float16* __restrict__ WT,
    const float* __restrict__ bias,
    _Float16* __restrict__ out_h, float* __restrict__ out_f, int mode) {
    __shared__ _Float16 XS[16 * CC];              // 24 KB A-strip
    const int tid = threadIdx.x;
    const int wave = tid >> 5, lane = tid & 31;
    const int half = lane >> 4, nl = lane & 15;
    const int m0 = blockIdx.x * 16;

    // ---- async stage A strip: 24576 contiguous bytes from X + m0*CC -------
    {
        const unsigned long long gx =
            (unsigned long long)(uintptr_t)(X + (size_t)m0 * CC);
        const unsigned lbase = (unsigned)(uintptr_t)XS + (unsigned)tid * 16u;
        const unsigned gbase = (unsigned)tid * 16u;
#pragma unroll
        for (int ch = 0; ch < 12; ++ch)           // 128 thr * 16 B * 12 = 24576
            async_copy_b128(lbase + ch * 2048u, gbase + ch * 2048u, gx);
        async_drain();
        __syncthreads();
    }

    const int nb = blockIdx.y * 256 + wave * 64;  // wave's 64-wide N strip
    v8f acc[4] = {};
    const _Float16* xr = XS + nl * CC;
    const _Float16* w0 = WT + (size_t)(nb +  0 + nl) * CC;
    const _Float16* w1 = WT + (size_t)(nb + 16 + nl) * CC;
    const _Float16* w2 = WT + (size_t)(nb + 32 + nl) * CC;
    const _Float16* w3 = WT + (size_t)(nb + 48 + nl) * CC;
#pragma unroll 2
    for (int kb = 0; kb < CC; kb += 32) {
        v16h a = load_a(xr + kb + half * 16, xr + kb + half * 16 + 8);
        acc[0] = wmma16x16x32(a, *(const v16h*)(w0 + kb + half * 16), acc[0]);
        acc[1] = wmma16x16x32(a, *(const v16h*)(w1 + kb + half * 16), acc[1]);
        acc[2] = wmma16x16x32(a, *(const v16h*)(w2 + kb + half * 16), acc[2]);
        acc[3] = wmma16x16x32(a, *(const v16h*)(w3 + kb + half * 16), acc[3]);
    }

#pragma unroll
    for (int t2 = 0; t2 < 4; ++t2) {
        const int bcol = nb + t2 * 16 + nl;
        const float bv = bias[bcol];
        const int h = bcol >> 6, d = bcol & 63;
#pragma unroll
        for (int r = 0; r < 8; ++r) {
            const int m = m0 + r + 8 * half;      // C/D: VGPR r -> M=r (+8 hi)
            const float val = acc[t2][r] + bv;
            const int b_ = m >> 10, t = m & (TT - 1);
            if (mode == 0) {
                out_h[(((size_t)(b_ * HH + h) * TT) + t) * DD + d] = (_Float16)val;
            } else if (mode == 2) {
                out_h[(((size_t)(b_ * HH + h) * DD) + d) * TT + t] = (_Float16)val;
            } else {
                out_f[(size_t)m * CC + bcol] = val;
            }
        }
    }
}

// ---------------------------------------------------------------------------
// Attention + UCB top-k + probs*V, one (b,h, 16-query tile) per block.
// grid = (B*H, T/16) = (48, 64), block = 256 (8 waves), dyn LDS = 16*1024*4 B.
// q tile (2 KB) async-staged to LDS once; A-operands hoisted out of key loop.
// ---------------------------------------------------------------------------
__global__ __launch_bounds__(256) void attn_kernel(
    const _Float16* __restrict__ q16,   // [B,H,T,D]
    const _Float16* __restrict__ k16,   // [B,H,T,D]
    const _Float16* __restrict__ vT,    // [B,H,D,T]
    const float* __restrict__ cnt_in,   // [B,H,T,T]
    float* __restrict__ probs_out,      // [B,H,T,T]
    float* __restrict__ cnt_out,        // [B,H,T,T]
    _Float16* __restrict__ ctx16,       // [B,T,C] f16
    const int* __restrict__ counter_p, const int* __restrict__ ucb_p) {
    __shared__ _Float16 QS[16 * DD];    // 2 KB q tile
    extern __shared__ float S[];        // [16][1024] score/att/prob tile
    const int bh = blockIdx.x;
    const int b = bh / HH, h = bh - b * HH;
    const int q0 = blockIdx.y * 16;
    const int wave = threadIdx.x >> 5, lane = threadIdx.x & 31;
    const int half = lane >> 4, nl = lane & 15;

    const _Float16* qbase = q16 + ((size_t)bh * TT + q0) * DD;
    const _Float16* kbase = k16 + (size_t)bh * TT * DD;

    // hint: count rows for this tile will stream through soon
    __builtin_prefetch(cnt_in + ((size_t)bh * TT + q0 + wave) * TT + lane * 4, 0, 0);

    // ---- async stage q tile (2048 contiguous bytes) ------------------------
    {
        if (wave == 0) {
            const unsigned long long gq = (unsigned long long)(uintptr_t)qbase;
            const unsigned lbase = (unsigned)(uintptr_t)QS + (unsigned)lane * 16u;
            const unsigned gbase = (unsigned)lane * 16u;
#pragma unroll
            for (int ch = 0; ch < 4; ++ch)        // 32 lanes * 16 B * 4 = 2048
                async_copy_b128(lbase + ch * 512u, gbase + ch * 512u, gq);
        }
        async_drain();
        __syncthreads();
    }

    // ---- Phase A: S = (q . k^T) / sqrt(D) via WMMA -------------------------
    // A-operands are loop-invariant: hoist both K-chunks into registers.
    const _Float16* qrowp = QS + nl * DD;
    const v16h a0 = load_a(qrowp + half * 16,      qrowp + half * 16 + 8);
    const v16h a1 = load_a(qrowp + 32 + half * 16, qrowp + 40 + half * 16);
    for (int jt = wave; jt < TT / 16; jt += 8) {
        const int key = jt * 16 + nl;
        const _Float16* kp = kbase + (size_t)key * DD;
        v16h b0 = *(const v16h*)(kp + half * 16);
        v16h b1 = *(const v16h*)(kp + 32 + half * 16);
        v8f c = {};
        c = wmma16x16x32(a0, b0, c);
        c = wmma16x16x32(a1, b1, c);
#pragma unroll
        for (int r = 0; r < 8; ++r)
            S[(r + 8 * half) * TT + jt * 16 + nl] = c[r] * 0.125f; // 1/sqrt(64)
    }
    __syncthreads();

    // ---- Phase B/C: softmax rows + UCB top-10 ------------------------------
    const bool ucb_on = (ucb_p[0] != 0) && (counter_p[0] >= 1000);
    const float log_t = logf((float)counter_p[0]);

    for (int rr = 0; rr < 2; ++rr) {
        const int r = wave + rr * 8;
        const int qrow = q0 + r;
        float* srow = S + r * TT;
        const size_t obase = ((size_t)bh * TT + qrow) * TT;
        const float* crow = cnt_in + obase;

        float av[32];
        float mx = -INFINITY;
#pragma unroll
        for (int i = 0; i < 32; ++i) { av[i] = srow[i * 32 + lane]; mx = fmaxf(mx, av[i]); }
        mx = wred_max(mx);
        float sum = 0.f;
#pragma unroll
        for (int i = 0; i < 32; ++i) { av[i] = __expf(av[i] - mx); sum += av[i]; }
        sum = wred_sum(sum);
        const float inv = 1.0f / sum;
#pragma unroll
        for (int i = 0; i < 32; ++i) av[i] *= inv;   // av = softmax(att)

        if (ucb_on) {
            float cv[32], uv[32];
#pragma unroll
            for (int i = 0; i < 32; ++i) cv[i] = crow[i * 32 + lane];
#pragma unroll
            for (int i = 0; i < 32; ++i)
                uv[i] = av[i] + sqrtf(log_t / (cv[i] + 1e-8f));

            int tj[KTOP];
#pragma unroll
            for (int t = 0; t < KTOP; ++t) {
                float bv = -INFINITY; int bj = 0x7fffffff;
#pragma unroll
                for (int i = 0; i < 32; ++i) {
                    const int j = i * 32 + lane;
                    if (uv[i] > bv || (uv[i] == bv && j < bj)) { bv = uv[i]; bj = j; }
                }
#pragma unroll
                for (int o = 1; o < 32; o <<= 1) {
                    const float ov = __shfl_xor(bv, o, 32);
                    const int oj = __shfl_xor(bj, o, 32);
                    if (ov > bv || (ov == bv && oj < bj)) { bv = ov; bj = oj; }
                }
                tj[t] = bj;
                if (lane == (bj & 31)) uv[bj >> 5] = -INFINITY;
            }

            float ssum = 0.f;
            bool sel[32];
#pragma unroll
            for (int i = 0; i < 32; ++i) {
                const int j = i * 32 + lane;
                bool s = false;
#pragma unroll
                for (int t = 0; t < KTOP; ++t) s = s || (tj[t] == j);
                sel[i] = s;
                if (s) ssum += av[i];
            }
            ssum = wred_sum(ssum);
            const float idn = 1.0f / (ssum + 1e-8f);
#pragma unroll
            for (int i = 0; i < 32; ++i) {
                const int j = i * 32 + lane;
                const float p = sel[i] ? av[i] * idn : 0.0f;
                probs_out[obase + j] = p;
                cnt_out[obase + j] = cv[i] + (sel[i] ? 1.0f : 0.0f);
                srow[j] = p;
            }
        } else {
#pragma unroll
            for (int i = 0; i < 32; ++i) {
                const int j = i * 32 + lane;
                probs_out[obase + j] = av[i];
                cnt_out[obase + j] = crow[j];
                srow[j] = av[i];
            }
        }
    }
    __syncthreads();

    // ---- Phase D: ctx[16,64] = probs[16,1024] * v[1024,64] via WMMA --------
    if (wave < 4) {
        const int d0 = wave * 16;
        const _Float16* vb = vT + ((size_t)bh * DD + d0 + nl) * TT;
        v8f c = {};
        for (int kb = 0; kb < TT; kb += 32) {
            v16h a;
#pragma unroll
            for (int e = 0; e < 8; ++e) {
                a[e]     = (_Float16)S[nl * TT + kb + half * 16 + e];
                a[e + 8] = (_Float16)S[nl * TT + kb + half * 16 + 8 + e];
            }
            v16h bm = *(const v16h*)(vb + kb + half * 16);
            c = wmma16x16x32(a, bm, c);
        }
#pragma unroll
        for (int r = 0; r < 8; ++r) {
            const int t = q0 + r + 8 * half;
            ctx16[((size_t)b * TT + t) * CC + h * DD + d0 + nl] = (_Float16)c[r];
        }
    }
}

// ---------------------------------------------------------------------------
// Host launcher
// ---------------------------------------------------------------------------
extern "C" void kernel_launch(void* const* d_in, const int* in_sizes, int n_in,
                              void* d_out, int out_size, void* d_ws, size_t ws_size,
                              hipStream_t stream) {
    (void)in_sizes; (void)n_in; (void)out_size; (void)ws_size;

    const float* hs   = (const float*)d_in[0];   // [B,T,C]
    const float* cnt  = (const float*)d_in[1];   // [B,H,T,T]
    const float* Wq   = (const float*)d_in[2];
    const float* bq   = (const float*)d_in[3];
    const float* Wk   = (const float*)d_in[4];
    const float* bk   = (const float*)d_in[5];
    const float* Wv   = (const float*)d_in[6];
    const float* bv   = (const float*)d_in[7];
    const float* Wo   = (const float*)d_in[8];
    const float* bo   = (const float*)d_in[9];
    const int* counter = (const int*)d_in[10];
    const int* ucb     = (const int*)d_in[11];

    float* out_f     = (float*)d_out;                              // [B,T,C]
    float* probs_out = out_f + (size_t)BB * TT * CC;               // [B,H,T,T]
    float* cnt_out   = probs_out + (size_t)BB * HH * TT * TT;      // [B,H,T,T]

    // workspace carve-up (all f16, 2-byte elements, 32B-aligned chunks)
    const size_t NTOK = (size_t)BB * TT;       // 4096
    _Float16* h16  = (_Float16*)d_ws;          // [4096,768]
    _Float16* wqT  = h16  + NTOK * CC;         // [768,768] transposed
    _Float16* wkT  = wqT  + (size_t)CC * CC;
    _Float16* wvT  = wkT  + (size_t)CC * CC;
    _Float16* woT  = wvT  + (size_t)CC * CC;
    _Float16* q16  = woT  + (size_t)CC * CC;   // [B,H,T,D]
    _Float16* k16  = q16  + NTOK * CC;
    _Float16* vT16 = k16  + NTOK * CC;         // [B,H,D,T]
    _Float16* ctx16 = vT16 + NTOK * CC;        // [B,T,C]

    // 1) precision conversion / weight transpose
    {
        const int n = (int)(NTOK * CC);
        cvt_f16<<<(n + 255) / 256, 256, 0, stream>>>(hs, h16, n);
        const int nw = CC * CC;
        const int g = (nw + 255) / 256;
        cvt_f16_T<<<g, 256, 0, stream>>>(Wq, wqT, CC, CC);
        cvt_f16_T<<<g, 256, 0, stream>>>(Wk, wkT, CC, CC);
        cvt_f16_T<<<g, 256, 0, stream>>>(Wv, wvT, CC, CC);
        cvt_f16_T<<<g, 256, 0, stream>>>(Wo, woT, CC, CC);
    }

    // 2) QKV projections (WMMA + async LDS staging)
    {
        dim3 grid(NTOK / 16, CC / 256);
        gemm_f16<<<grid, 128, 0, stream>>>(h16, wqT, bq, q16,  nullptr, 0);
        gemm_f16<<<grid, 128, 0, stream>>>(h16, wkT, bk, k16,  nullptr, 0);
        gemm_f16<<<grid, 128, 0, stream>>>(h16, wvT, bv, vT16, nullptr, 2);
    }

    // 3) attention + UCB top-k + probs*V   (64 KB dynamic LDS score tile)
    {
        dim3 grid(BB * HH, TT / 16);
        attn_kernel<<<grid, 256, 16 * TT * sizeof(float), stream>>>(
            q16, k16, vT16, cnt, probs_out, cnt_out, ctx16, counter, ucb);
    }

    // 4) output projection (WMMA, f32 result into d_out)
    {
        dim3 grid(NTOK / 16, CC / 256);
        gemm_f16<<<grid, 128, 0, stream>>>(ctx16, woT, bo, nullptr, out_f, 3);
    }
}
